// NeuralGraph_35081292874420
// MI455X (gfx1250) — matrix operand order repaired and checked by reading, not compiled, not run on hardware
//
#include <hip/hip_runtime.h>
#include <cmath>

static constexpr int   B_     = 2;
static constexpr int   NN     = 100000;
static constexpr int   NE     = 1600000;
static constexpr float MAXV   = 1000000.0f;
static constexpr float SCALE  = 0.35355339059327373f; // 1/sqrt(8)

typedef __attribute__((ext_vector_type(16))) _Float16 v16h;
typedef __attribute__((ext_vector_type(8)))  float    v8f;

__device__ __forceinline__ v8f wmma16(v16h a, v16h b, v8f c) {
  // D = A(16x32 f16) * B(32x16 f16) + C(16x16 f32)
  return __builtin_amdgcn_wmma_f32_16x16x32_f16(false, a, false, b, (short)0, c, false, false);
}

__device__ __forceinline__ v8f v8f_zero() {
  v8f z;
#pragma unroll
  for (int i = 0; i < 8; ++i) z[i] = 0.f;
  return z;
}

// Branch-free fast tanh on the native trans unit:
// tanh(x) = sign(x) * (1 - e^{-2|x|}) / (1 + e^{-2|x|})
__device__ __forceinline__ float fast_tanh(float x) {
  const float ax = __builtin_fabsf(x);
  // e^{-2ax} = exp2(-2*log2(e) * ax)
  const float t = __builtin_amdgcn_exp2f(ax * -2.8853900817779268f);
  const float r = (1.0f - t) * __builtin_amdgcn_rcpf(1.0f + t);
  return __builtin_copysignf(r, x);
}

// exp for non-positive arguments (softmax: logit - max <= 0), no range handling needed.
__device__ __forceinline__ float fast_exp_np(float x) {
  return __builtin_amdgcn_exp2f(x * 1.4426950408889634f);
}

__device__ __forceinline__ float soft_clamp(float x) {
  return MAXV * fast_tanh(x * (1.0f / MAXV));
}

// Monotonic-int float atomic max; buffer must be initialized to -inf bits.
__device__ __forceinline__ void atomicMaxF(float* addr, float v) {
  int iv = __float_as_int(v);
  if (iv >= 0) atomicMax((int*)addr, iv);
  else         atomicMin((unsigned int*)addr, (unsigned int)iv);
}

// ---------------- init workspace ----------------
__global__ void k_init(float* maxa, float* maxb, float* dena, float* denb,
                       float* numa, float* numb) {
  int i = blockIdx.x * blockDim.x + threadIdx.x;
  if (i >= B_ * NN) return;
  float ninf = __int_as_float(0xff800000u);
  maxa[i] = ninf; maxb[i] = ninf;
  dena[i] = 0.f;  denb[i] = 0.f;
#pragma unroll
  for (int c = 0; c < 8; ++c) { numa[(size_t)i * 8 + c] = 0.f; numb[(size_t)i * 8 + c] = 0.f; }
}

// ---------------- per-node attention MLP 8->16->32 ----------------
__global__ void k_att(const float* __restrict__ nodes,
                      const float* __restrict__ w1, const float* __restrict__ b1,
                      const float* __restrict__ w2, const float* __restrict__ b2,
                      float* __restrict__ att) {
  int i = blockIdx.x * blockDim.x + threadIdx.x;
  if (i >= B_ * NN) return;
  float x[8];
#pragma unroll
  for (int c = 0; c < 8; ++c) x[c] = nodes[(size_t)i * 8 + c];
  float h[16];
#pragma unroll
  for (int j = 0; j < 16; ++j) {
    float s = b1[j];
#pragma unroll
    for (int c = 0; c < 8; ++c) s += x[c] * w1[c * 16 + j];
    h[j] = fast_tanh(s);
  }
#pragma unroll
  for (int o = 0; o < 32; ++o) {
    float s = b2[o];
#pragma unroll
    for (int j = 0; j < 16; ++j) s += h[j] * w2[j * 32 + o];
    att[(size_t)i * 32 + o] = s;
  }
}

// ---------------- per-edge message MLP (WMMA) + logits + max-pass ----------------
__global__ __launch_bounds__(256) void k_msg(
    const float* __restrict__ nodes, const float* __restrict__ edges,
    const float* __restrict__ w1, const float* __restrict__ b1,
    const float* __restrict__ w2, const float* __restrict__ b2,
    const int* __restrict__ src, const int* __restrict__ tgt,
    const float* __restrict__ att,
    float* __restrict__ ma, float* __restrict__ mb,
    float* __restrict__ la, float* __restrict__ lb,
    float* __restrict__ maxa, float* __restrict__ maxb,
    float* __restrict__ out_edges) {
  __shared__ float lds[8][16][17];
  const int lane = threadIdx.x & 31;
  const int wave = threadIdx.x >> 5;
  const int row  = lane & 15;
  const int hi   = lane >> 4;

  // B-operands (loop invariant). ISA 32x16 f16 B layout: half h <-> K = (h<8 ? hi*8+h : 16+hi*8+h-8), N = row.
  v16h bw1, bw2lo, bw2hi;
#pragma unroll
  for (int h = 0; h < 16; ++h) {
    int k = (h < 8) ? hi * 8 + h : 16 + hi * 8 + (h - 8);
    bw1[h]   = (k < 24)           ? (_Float16)w1[k * 16 + row]      : (_Float16)0.f;
    bw2lo[h] = (k < 16)           ? (_Float16)w2[k * 24 + row]      : (_Float16)0.f;
    bw2hi[h] = (k < 16 && row < 8) ? (_Float16)w2[k * 24 + 16 + row] : (_Float16)0.f;
  }
  const float b1v   = b1[row];
  const float b2lov = b2[row];
  const float b2hiv = (row < 8) ? b2[16 + row] : 0.f;

  const int tile = blockIdx.x * 8 + wave;        // grid sized exactly: always valid
  const int b    = tile / (NE / 16);
  const int et   = tile % (NE / 16);
  const int e    = et * 16 + row;
  const size_t ge = (size_t)b * NE + e;

  const int s_idx = src[e];
  const int t_idx = tgt[e];

  // A-operand: x = [nodes[src](0..7), nodes[tgt](8..15), edges(16..23), pad(24..31)]
  // ISA 16x32 f16 A layout: lane-lo holds K{0..7,16..23}, lane-hi K{8..15,24..31}.
  v16h a;
  {
    const float* nb = nodes + ((size_t)b * NN + (hi ? t_idx : s_idx)) * 8;
#pragma unroll
    for (int h = 0; h < 8; ++h) a[h] = (_Float16)nb[h];
    if (!hi) {
      const float* eb = edges + ge * 8;
#pragma unroll
      for (int h = 0; h < 8; ++h) a[8 + h] = (_Float16)eb[h];
    } else {
#pragma unroll
      for (int h = 0; h < 8; ++h) a[8 + h] = (_Float16)0.f;
    }
  }

  v8f c1 = wmma16(a, bw1, v8f_zero());
  // hidden (16 edges x 16) -> LDS for operand transpose into GEMM2's A
#pragma unroll
  for (int r = 0; r < 8; ++r) lds[wave][r + 8 * hi][row] = fast_tanh(c1[r] + b1v);
  __syncthreads();
  v16h a2;
#pragma unroll
  for (int h = 0; h < 8; ++h) {
    a2[h]     = (_Float16)lds[wave][row][hi * 8 + h];
    a2[8 + h] = (_Float16)0.f;
  }

  v8f dlo = wmma16(a2, bw2lo, v8f_zero());   // m[:, 0:16]
  v8f dhi = wmma16(a2, bw2hi, v8f_zero());   // m[:, 16:24]

#pragma unroll
  for (int r = 0; r < 8; ++r) {
    const int    er = et * 16 + r + 8 * hi;
    const size_t g  = ((size_t)b * NE + er) * 8;
    const float mlo = dlo[r] + b2lov;
    if (row < 8) {
      ma[g + row] = mlo;                                      // m_a
      const float ev = edges[g + row];
      out_edges[g + row] = soft_clamp(ev + dhi[r] + b2hiv);   // edges + m_ab
    } else {
      mb[g + row - 8] = mlo;                                  // m_b
    }
  }

  // logits: lanes 0-15 -> logit_a, lanes 16-31 -> logit_b (same 16 edges)
  {
    const float* as_ = att + ((size_t)b * NN + s_idx) * 32;
    const float* at_ = att + ((size_t)b * NN + t_idx) * 32;
    const float* q  = hi ? (at_ + 24) : (as_ + 8);   // b_query[tgt] : a_query[src]
    const float* kk = hi ? (as_ + 0)  : (at_ + 16);  // a_key[src]   : b_key[tgt]
    float lg = 0.f;
#pragma unroll
    for (int i = 0; i < 8; ++i) lg += q[i] * kk[i];
    lg *= SCALE;
    const int seg = hi ? t_idx : s_idx;
    float* mx = hi ? maxb : maxa;
    float* lo = hi ? lb : la;
    atomicMaxF(&mx[b * NN + seg], lg);
    lo[ge] = lg;
  }
}

// ---------------- softmax accumulation pass ----------------
__global__ void k_acc(const int* __restrict__ src, const int* __restrict__ tgt,
                      const float* __restrict__ la, const float* __restrict__ lb,
                      const float* __restrict__ ma, const float* __restrict__ mb,
                      const float* __restrict__ maxa, const float* __restrict__ maxb,
                      float* __restrict__ dena, float* __restrict__ denb,
                      float* __restrict__ numa, float* __restrict__ numb) {
  size_t i = (size_t)blockIdx.x * blockDim.x + threadIdx.x;
  if (i >= (size_t)B_ * NE) return;
  int b = (int)(i / NE), e = (int)(i % NE);
  int s = src[e], t = tgt[e];
  size_t sa = (size_t)b * NN + s, sb = (size_t)b * NN + t;
  float ea = fast_exp_np(la[i] - maxa[sa]);   // arg <= 0 by construction
  float eb = fast_exp_np(lb[i] - maxb[sb]);
  atomicAdd(&dena[sa], ea);
  atomicAdd(&denb[sb], eb);
#pragma unroll
  for (int c = 0; c < 8; ++c) {
    atomicAdd(&numa[sa * 8 + c], ea * ma[i * 8 + c]);
    atomicAdd(&numb[sb * 8 + c], eb * mb[i * 8 + c]);
  }
}

// ---------------- per-node update MLP (WMMA) ----------------
__global__ __launch_bounds__(256) void k_upd(
    const float* __restrict__ nodes,
    const float* __restrict__ w1, const float* __restrict__ b1,
    const float* __restrict__ w2, const float* __restrict__ b2,
    const float* __restrict__ dena, const float* __restrict__ denb,
    const float* __restrict__ numa, const float* __restrict__ numb,
    float* __restrict__ out_nodes) {
  __shared__ float lds[8][16][17];
  const int lane = threadIdx.x & 31;
  const int wave = threadIdx.x >> 5;
  const int row  = lane & 15;
  const int hi   = lane >> 4;

  const int n_tiles = B_ * (NN / 16);
  int tile = blockIdx.x * 8 + wave;
  const bool valid = tile < n_tiles;
  if (!valid) tile = 0;
  const int b  = tile / (NN / 16);
  const int nt = tile % (NN / 16);
  const int node = nt * 16 + row;
  const size_t gn = (size_t)b * NN + node;

  v16h bw1, bw2;
#pragma unroll
  for (int h = 0; h < 16; ++h) {
    int k = (h < 8) ? hi * 8 + h : 16 + hi * 8 + (h - 8);
    bw1[h] = (k < 24)          ? (_Float16)w1[k * 16 + row] : (_Float16)0.f;
    bw2[h] = (k < 16 && row < 8) ? (_Float16)w2[k * 8 + row]  : (_Float16)0.f;
  }
  const float b1v = b1[row];
  const float b2v = (row < 8) ? b2[row] : 0.f;

  // u_x = [nodes(0..7), agg_a(8..15), agg_b(16..23), pad]
  v16h a;
  if (!hi) {
    float rd = __builtin_amdgcn_rcpf(denb[gn] + 1e-10f);
#pragma unroll
    for (int h = 0; h < 8; ++h) {
      a[h]     = (_Float16)nodes[gn * 8 + h];
      a[8 + h] = (_Float16)(numb[gn * 8 + h] * rd);   // K=16..23 -> agg_b
    }
  } else {
    float rd = __builtin_amdgcn_rcpf(dena[gn] + 1e-10f);
#pragma unroll
    for (int h = 0; h < 8; ++h) {
      a[h]     = (_Float16)(numa[gn * 8 + h] * rd);   // K=8..15 -> agg_a
      a[8 + h] = (_Float16)0.f;
    }
  }

  v8f c1 = wmma16(a, bw1, v8f_zero());
#pragma unroll
  for (int r = 0; r < 8; ++r) lds[wave][r + 8 * hi][row] = fast_tanh(c1[r] + b1v);
  __syncthreads();
  v16h a2;
#pragma unroll
  for (int h = 0; h < 8; ++h) {
    a2[h]     = (_Float16)lds[wave][row][hi * 8 + h];
    a2[8 + h] = (_Float16)0.f;
  }
  v8f d = wmma16(a2, bw2, v8f_zero());

  if (valid && row < 8) {
#pragma unroll
    for (int r = 0; r < 8; ++r) {
      int nr = nt * 16 + r + 8 * hi;
      size_t g = ((size_t)b * NN + nr) * 8;
      float nv = nodes[g + row];
      out_nodes[g + row] = soft_clamp(nv + d[r] + b2v);
    }
  }
}

extern "C" void kernel_launch(void* const* d_in, const int* in_sizes, int n_in,
                              void* d_out, int out_size, void* d_ws, size_t ws_size,
                              hipStream_t stream) {
  (void)in_sizes; (void)n_in; (void)out_size; (void)ws_size;
  const float* nodes  = (const float*)d_in[0];
  const float* edges  = (const float*)d_in[1];
  const float* msg_w1 = (const float*)d_in[2];
  const float* msg_b1 = (const float*)d_in[3];
  const float* msg_w2 = (const float*)d_in[4];
  const float* msg_b2 = (const float*)d_in[5];
  const float* att_w1 = (const float*)d_in[6];
  const float* att_b1 = (const float*)d_in[7];
  const float* att_w2 = (const float*)d_in[8];
  const float* att_b2 = (const float*)d_in[9];
  const float* upd_w1 = (const float*)d_in[10];
  const float* upd_b1 = (const float*)d_in[11];
  const float* upd_w2 = (const float*)d_in[12];
  const float* upd_b2 = (const float*)d_in[13];
  const int* sources  = (const int*)d_in[14];
  const int* targets  = (const int*)d_in[15];

  float* out_nodes = (float*)d_out;
  float* out_edges = out_nodes + (size_t)B_ * NN * 8;

  float* ws = (float*)d_ws;
  size_t o = 0;
  float* att  = ws + o; o += (size_t)B_ * NN * 32;
  float* la   = ws + o; o += (size_t)B_ * NE;
  float* lb   = ws + o; o += (size_t)B_ * NE;
  float* ma   = ws + o; o += (size_t)B_ * NE * 8;
  float* mb   = ws + o; o += (size_t)B_ * NE * 8;
  float* maxa = ws + o; o += (size_t)B_ * NN;
  float* maxb = ws + o; o += (size_t)B_ * NN;
  float* dena = ws + o; o += (size_t)B_ * NN;
  float* denb = ws + o; o += (size_t)B_ * NN;
  float* numa = ws + o; o += (size_t)B_ * NN * 8;
  float* numb = ws + o; o += (size_t)B_ * NN * 8;

  const int bn = B_ * NN;
  k_init<<<(bn + 255) / 256, 256, 0, stream>>>(maxa, maxb, dena, denb, numa, numb);
  k_att<<<(bn + 255) / 256, 256, 0, stream>>>(nodes, att_w1, att_b1, att_w2, att_b2, att);
  k_msg<<<(B_ * (NE / 16)) / 8, 256, 0, stream>>>(nodes, edges, msg_w1, msg_b1, msg_w2,
                                                  msg_b2, sources, targets, att, ma, mb,
                                                  la, lb, maxa, maxb, out_edges);
  k_acc<<<(B_ * NE + 255) / 256, 256, 0, stream>>>(sources, targets, la, lb, ma, mb,
                                                   maxa, maxb, dena, denb, numa, numb);
  const int n_tiles = B_ * (NN / 16);
  k_upd<<<(n_tiles + 7) / 8, 256, 0, stream>>>(nodes, upd_w1, upd_b1, upd_w2, upd_b2,
                                               dena, denb, numa, numb, out_nodes);
}
